// StructureHead_60455959659073
// MI455X (gfx1250) — compile-verified, matrix operand order checked by reading, not run on hardware
//
#include <hip/hip_runtime.h>
#include <hip/hip_bf16.h>
#include <math.h>

#define N_RES 512
#define CS    384
#define CZ    128
#define NH    12
#define CH    16
#define NPQ   4
#define NPV   8
#define NBLK  8
#define PROJD 1152   // 192 q + 384 kv + 144 qp + 432 kvp
#define CATD  2112
#define QP_OFF  576
#define KVP_OFF 720

typedef __attribute__((ext_vector_type(16))) _Float16 v16h;
typedef __attribute__((ext_vector_type(8)))  float    v8f;

// ---------------------------------------------------------------- WMMA helpers
__device__ __forceinline__ v8f wmma32(v16h a, v16h b, v8f c) {
#if defined(__HIP_DEVICE_COMPILE__)
  return __builtin_amdgcn_wmma_f32_16x16x32_f16(false, a, false, b, (short)0, c, false, false);
#else
  return c;
#endif
}

// A/B tile loader, contiguous K (row-major over k). Layout per CDNA5 ISA 7.12.2:
// lanes 0-15: elems 0..7 = K k0+0..7, elems 8..15 = K k0+16..23
// lanes 16-31: elems 0..7 = K k0+8..15, elems 8..15 = K k0+24..31
__device__ __forceinline__ void ld_k32(const _Float16* __restrict__ rowp, int lane, v16h& d) {
  const _Float16* p = rowp + ((lane >> 4) << 3);
#pragma unroll
  for (int e = 0; e < 8; ++e) { d[e] = p[e]; d[e + 8] = p[e + 16]; }
}
// same but K is strided (source is [k][n] row-major over n)
__device__ __forceinline__ void ld_k32_strided(const _Float16* __restrict__ base, int ldn, int lane, v16h& d) {
  const _Float16* p = base + (size_t)((lane >> 4) << 3) * ldn;
#pragma unroll
  for (int e = 0; e < 8; ++e) { d[e] = p[(size_t)e * ldn]; d[e + 8] = p[(size_t)(e + 16) * ldn]; }
}

__device__ __forceinline__ float wave_sum(float v) {
#pragma unroll
  for (int m = 16; m > 0; m >>= 1) v += __shfl_xor(v, m, 32);
  return v;
}
__device__ __forceinline__ float wave_max(float v) {
#pragma unroll
  for (int m = 16; m > 0; m >>= 1) v = fmaxf(v, __shfl_xor(v, m, 32));
  return v;
}

__device__ __forceinline__ void quat_to_rot(const float q[4], float R[3][3]) {
  float n = rsqrtf(q[0]*q[0]+q[1]*q[1]+q[2]*q[2]+q[3]*q[3] + 1e-30f);
  float w = q[0]*n, x = q[1]*n, y = q[2]*n, z = q[3]*n;
  R[0][0] = 1.f-2.f*(y*y+z*z); R[0][1] = 2.f*(x*y-w*z); R[0][2] = 2.f*(x*z+w*y);
  R[1][0] = 2.f*(x*y+w*z); R[1][1] = 1.f-2.f*(x*x+z*z); R[1][2] = 2.f*(y*z-w*x);
  R[2][0] = 2.f*(x*z-w*y); R[2][1] = 2.f*(y*z+w*x); R[2][2] = 1.f-2.f*(x*x+y*y);
}

// ---------------------------------------------------------------- small utils
__global__ void k_f32_to_f16(const float* __restrict__ a, _Float16* __restrict__ o, int n) {
  int i = blockIdx.x * blockDim.x + threadIdx.x;
  if (i < n) o[i] = (_Float16)a[i];
}
__global__ void k_copy_f32(const float* __restrict__ a, float* __restrict__ o, int n) {
  int i = blockIdx.x * blockDim.x + threadIdx.x;
  if (i < n) o[i] = a[i];
}
__global__ void k_init_rigid(const float* __restrict__ rq, const float* __restrict__ rt,
                             float* __restrict__ quats, float* __restrict__ trans) {
  int i = blockIdx.x * blockDim.x + threadIdx.x;
  if (i >= N_RES) return;
  float q0 = rq[i*4+0], q1 = rq[i*4+1], q2 = rq[i*4+2], q3 = rq[i*4+3];
  float inv = rsqrtf(q0*q0+q1*q1+q2*q2+q3*q3 + 1e-30f);
  quats[i*4+0] = q0*inv; quats[i*4+1] = q1*inv; quats[i*4+2] = q2*inv; quats[i*4+3] = q3*inv;
  trans[i*3+0] = rt[i*3+0]*0.1f; trans[i*3+1] = rt[i*3+1]*0.1f; trans[i*3+2] = rt[i*3+2]*0.1f;
}

// ---------------------------------------------------------------- LN(z) + bpair (once)
__global__ __launch_bounds__(256) void k_lnz_bpair(const float* __restrict__ z,
    const float* __restrict__ g, const float* __restrict__ b,
    const float* __restrict__ wb, const float* __restrict__ bb,
    _Float16* __restrict__ z16, float* __restrict__ bpair) {
  int lane = threadIdx.x & 31, wave = threadIdx.x >> 5;
  int row = blockIdx.x * 8 + wave;                   // i*512 + j
  const float* zr = z + (size_t)row * CZ;
  float x[4]; float s = 0.f;
#pragma unroll
  for (int e = 0; e < 4; ++e) { x[e] = zr[lane*4+e]; s += x[e]; }
  float mean = wave_sum(s) * (1.f/CZ);
  float vs = 0.f;
#pragma unroll
  for (int e = 0; e < 4; ++e) { float d = x[e]-mean; vs += d*d; }
  float inv = rsqrtf(wave_sum(vs)*(1.f/CZ) + 1e-5f);
  float nv[4];
#pragma unroll
  for (int e = 0; e < 4; ++e) {
    int c = lane*4+e;
    nv[e] = (x[e]-mean)*inv*g[c] + b[c];
    z16[(size_t)row*CZ + c] = (_Float16)nv[e];
  }
  int ii = row >> 9, jj = row & 511;
  for (int h = 0; h < NH; ++h) {
    float p = 0.f;
#pragma unroll
    for (int e = 0; e < 4; ++e) p += nv[e]*wb[h*CZ + lane*4+e];
    p = wave_sum(p);
    if (lane == 0) bpair[((size_t)h*N_RES + ii)*N_RES + jj] = 0.57735027f*(p + bb[h]);
  }
}

// ---------------------------------------------------------------- generic LN rows (D=384)
__global__ __launch_bounds__(256) void k_ln_rows(const float* __restrict__ in,
    const float* __restrict__ g, const float* __restrict__ b,
    float* __restrict__ outF, _Float16* __restrict__ outH) {
  int lane = threadIdx.x & 31, wave = threadIdx.x >> 5;
  int row = blockIdx.x * 8 + wave;
  const float* r = in + (size_t)row * CS;
  float v[12]; float s = 0.f;
#pragma unroll
  for (int e = 0; e < 12; ++e) { v[e] = r[lane + e*32]; s += v[e]; }
  float mean = wave_sum(s) * (1.f/CS);
  float vs = 0.f;
#pragma unroll
  for (int e = 0; e < 12; ++e) { float d = v[e]-mean; vs += d*d; }
  float inv = rsqrtf(wave_sum(vs)*(1.f/CS) + 1e-5f);
#pragma unroll
  for (int e = 0; e < 12; ++e) {
    int c = lane + e*32;
    float y = (v[e]-mean)*inv*g[c] + b[c];
    if (outF) outF[(size_t)row*CS + c] = y;
    if (outH) outH[(size_t)row*CS + c] = (_Float16)y;
  }
}

// ---------------------------------------------------------------- generic WMMA GEMM: Out = A(f16) @ W^T + bias (+res)(+relu)
__global__ __launch_bounds__(256) void k_gemm(const _Float16* __restrict__ A, int lda,
    const _Float16* __restrict__ W, int ldw, const float* __restrict__ bias,
    const float* __restrict__ residual, float* __restrict__ outF,
    _Float16* __restrict__ outH, int ldo, int M, int Nd, int K, int relu) {
  int lane = threadIdx.x & 31, wave = threadIdx.x >> 5;
  int n0 = blockIdx.x * 128 + wave * 16;
  int m0 = blockIdx.y * 16;
  if (n0 >= Nd || m0 >= M) return;
  const _Float16* Ar = A + (size_t)(m0 + (lane & 15)) * lda;
  const _Float16* Wr = W + (size_t)(n0 + (lane & 15)) * ldw;
  v8f acc = {};
  for (int k0 = 0; k0 < K; k0 += 32) {
    v16h a, bm;
    ld_k32(Ar + k0, lane, a);
    ld_k32(Wr + k0, lane, bm);
    __builtin_prefetch(Ar + k0 + 128);
    acc = wmma32(a, bm, acc);
  }
  int n = n0 + (lane & 15);
  int rb = m0 + ((lane < 16) ? 0 : 8);
  float bv = bias ? bias[n] : 0.f;
#pragma unroll
  for (int r = 0; r < 8; ++r) {
    int m = rb + r;
    float v = acc[r] + bv;
    if (residual) v += residual[(size_t)m * ldo + n];
    if (relu) v = fmaxf(v, 0.f);
    if (outF) outF[(size_t)m * ldo + n] = v;
    if (outH) outH[(size_t)m * ldo + n] = (_Float16)v;
  }
}

// ---------------------------------------------------------------- point transforms + fused attention operands
__global__ void k_points(const float* __restrict__ proj, const float* __restrict__ quats,
    const float* __restrict__ trans, const float* __restrict__ head_w,
    _Float16* __restrict__ qhat, _Float16* __restrict__ khat, _Float16* __restrict__ vcat,
    float* __restrict__ ssq, float* __restrict__ ssk) {
  int idx = blockIdx.x * blockDim.x + threadIdx.x;
  if (idx >= N_RES * NH) return;
  int i = idx / NH, h = idx % NH;
  float q[4] = {quats[i*4], quats[i*4+1], quats[i*4+2], quats[i*4+3]};
  float R[3][3]; quat_to_rot(q, R);
  float t[3] = {trans[i*3], trans[i*3+1], trans[i*3+2]};
  float hw = logf(1.f + expf(head_w[h])) * 0.13608276f;   // softplus * sqrt(1/54)
  const float qscale = 0.14433757f;                       // sqrt(1/48)
  const float* pr = proj + (size_t)i * PROJD;
  _Float16* qh = qhat + ((size_t)h*N_RES + i) * 32;
  _Float16* kh = khat + ((size_t)h*N_RES + i) * 32;
  _Float16* vc = vcat + ((size_t)h*N_RES + i) * 48;
#pragma unroll
  for (int c = 0; c < CH; ++c) {
    qh[c] = (_Float16)(pr[h*CH + c] * qscale);
    kh[c] = (_Float16)pr[192 + h*32 + c];
    vc[c] = (_Float16)pr[192 + h*32 + 16 + c];
  }
  float sq = 0.f;
  for (int p = 0; p < NPQ; ++p) {
    float raw[3], pt[3];
    for (int x = 0; x < 3; ++x) raw[x] = pr[QP_OFF + x*48 + h*NPQ + p];
    for (int x = 0; x < 3; ++x) {
      pt[x] = R[x][0]*raw[0] + R[x][1]*raw[1] + R[x][2]*raw[2] + t[x];
      qh[16 + p*3 + x] = (_Float16)(pt[x]*hw);
    }
    sq += pt[0]*pt[0]+pt[1]*pt[1]+pt[2]*pt[2];
  }
  float sk = 0.f;
  for (int pp = 0; pp < NPQ + NPV; ++pp) {
    float raw[3], pt[3];
    for (int x = 0; x < 3; ++x) raw[x] = pr[KVP_OFF + x*144 + h*(NPQ+NPV) + pp];
    for (int x = 0; x < 3; ++x)
      pt[x] = R[x][0]*raw[0] + R[x][1]*raw[1] + R[x][2]*raw[2] + t[x];
    if (pp < NPQ) {
      for (int x = 0; x < 3; ++x) kh[16 + pp*3 + x] = (_Float16)pt[x];
      sk += pt[0]*pt[0]+pt[1]*pt[1]+pt[2]*pt[2];
    } else {
      for (int x = 0; x < 3; ++x) vc[16 + (pp-NPQ)*3 + x] = (_Float16)pt[x];
    }
  }
  for (int c = 28; c < 32; ++c) { qh[c] = (_Float16)0.f; kh[c] = (_Float16)0.f; }
  for (int c = 40; c < 48; ++c) vc[c] = (_Float16)0.f;
  ssq[h*N_RES + i] = -0.5f*hw*sq;
  ssk[h*N_RES + i] = -0.5f*hw*sk;
}

// ---------------------------------------------------------------- attention: logits (WMMA K=32) + softmax + o/o_pt (WMMA K=512)
__global__ __launch_bounds__(256) void k_attn(const _Float16* __restrict__ qhat,
    const _Float16* __restrict__ khat, const float* __restrict__ ssq,
    const float* __restrict__ ssk, const float* __restrict__ bpair,
    const float* __restrict__ mask, const _Float16* __restrict__ vcat,
    _Float16* __restrict__ aout, float* __restrict__ obuf, float* __restrict__ optbuf) {
  __shared__ float Lg[16][N_RES];    // 32 KB
  int lane = threadIdx.x & 31, wave = threadIdx.x >> 5;
  int h = blockIdx.y, i0 = blockIdx.x * 16;
  // phase 1: logits, each wave covers 64 columns
  v16h aq; ld_k32(qhat + ((size_t)h*N_RES + i0 + (lane & 15)) * 32, lane, aq);
  for (int jt = 0; jt < 4; ++jt) {
    int j0 = wave*64 + jt*16;
    v16h bk; ld_k32(khat + ((size_t)h*N_RES + j0 + (lane & 15)) * 32, lane, bk);
    v8f c = {};
    c = wmma32(aq, bk, c);
    int n = j0 + (lane & 15);
    int rb = (lane < 16) ? 0 : 8;
    float skj = ssk[h*N_RES + n];
    float mj = mask[n];
#pragma unroll
    for (int r = 0; r < 8; ++r) {
      int row = rb + r, ig = i0 + row;
      float v = c[r] + bpair[((size_t)h*N_RES + ig)*N_RES + n]
              + ssq[h*N_RES + ig] + skj + 100000.f*(mask[ig]*mj - 1.f);
      Lg[row][n] = v;
    }
  }
  __syncthreads();
  // phase 2: softmax, 2 rows per wave; emit f16 attention to global for o_pair pass
  for (int rr = 0; rr < 2; ++rr) {
    int row = wave*2 + rr, ig = i0 + row;
    float mx = -3.0e38f;
    for (int j = lane; j < N_RES; j += 32) mx = fmaxf(mx, Lg[row][j]);
    mx = wave_max(mx);
    float s = 0.f;
    for (int j = lane; j < N_RES; j += 32) { float e = __expf(Lg[row][j]-mx); Lg[row][j] = e; s += e; }
    s = wave_sum(s);
    float inv = 1.f / s;
    _Float16* ao = aout + ((size_t)h*N_RES + ig) * N_RES;
    for (int j = lane; j < N_RES; j += 32) { float a = Lg[row][j]*inv; Lg[row][j] = a; ao[j] = (_Float16)a; }
  }
  __syncthreads();
  // phase 3: [o | o_pt] = a @ vcat  (16 x 48), waves 0..2
  if (wave < 3) {
    int n0 = wave * 16;
    int n = n0 + (lane & 15);
    v8f acc = {};
    for (int k0 = 0; k0 < N_RES; k0 += 32) {
      v16h a, bm;
      int arow = lane & 15, kb = (lane >> 4) << 3;
#pragma unroll
      for (int e = 0; e < 8; ++e) {
        a[e]   = (_Float16)Lg[arow][k0 + kb + e];
        a[e+8] = (_Float16)Lg[arow][k0 + kb + 16 + e];
      }
      ld_k32_strided(vcat + ((size_t)h*N_RES + k0) * 48 + n, 48, lane, bm);
      acc = wmma32(a, bm, acc);
    }
    int rb = (lane < 16) ? 0 : 8;
#pragma unroll
    for (int r = 0; r < 8; ++r) {
      int ig = i0 + rb + r;
      float v = acc[r];
      if (n < 16) obuf[(size_t)ig*(NH*CH) + h*CH + n] = v;
      else if (n < 40) {
        int qq = n - 16;
        optbuf[(((size_t)ig*NH + h)*NPV + (qq/3))*3 + (qq%3)] = v;
      }
    }
  }
}

// ---------------------------------------------------------------- o_pair: per-residue (16 x 512)x(512 x 128) via LDS-staged z
#if defined(__HIP_DEVICE_COMPILE__) && defined(__gfx1250__) && defined(__has_builtin)
#if __has_builtin(__builtin_amdgcn_tensor_load_to_lds)
#define ENABLE_TDM 1
#endif
#endif

#ifdef ENABLE_TDM
typedef unsigned int tdm4u __attribute__((ext_vector_type(4)));
typedef int tdm8i __attribute__((ext_vector_type(8)));
typedef int tdm4i __attribute__((ext_vector_type(4)));
#endif

__global__ __launch_bounds__(256) void k_opair(const _Float16* __restrict__ aatt,
    const _Float16* __restrict__ z16, float* __restrict__ opair) {
  __shared__ uint4 zsh4[2048];       // 32 KB: 128 rows x 128 f16
  _Float16* zsh = (_Float16*)zsh4;
  int lane = threadIdx.x & 31, wave = threadIdx.x >> 5;
  int i = blockIdx.x;
  int n0 = wave * 16;
  int n = n0 + (lane & 15);
  v8f acc = {};
  for (int c = 0; c < 4; ++c) {
    __syncthreads();   // previous chunk fully consumed
    const _Float16* src = z16 + ((size_t)i*N_RES + c*128) * CZ;
#ifdef ENABLE_TDM
    if (threadIdx.x < 32) {   // wave 0 drives the Tensor Data Mover
      unsigned long long ga = (unsigned long long)(uintptr_t)src;
      unsigned lds = (unsigned)(size_t)(void*)zsh4;
      tdm4u g0;
      g0.x = 1u;                                            // count=1, load
      g0.y = lds;                                           // lds_addr
      g0.z = (unsigned)(ga & 0xFFFFFFFFu);                  // global_addr lo
      g0.w = (unsigned)((ga >> 32) & 0x1FFFFFFu) | (2u << 30); // addr hi | type=2
      tdm8i g1;
      g1[0] = (3 << 16);          // data_size = 8B units
      g1[1] = (int)(32u << 16);   // tensor_dim0 = 32 (8B units per row)
      g1[2] = (int)(128u << 16);  // tensor_dim1 = 128 rows
      g1[3] = (int)(32u << 16);   // tile_dim0 = 32
      g1[4] = 128;                // tile_dim1 = 128
      g1[5] = 32;                 // tensor_dim0_stride = 32
      g1[6] = 0; g1[7] = 0;
      tdm4i gz = {0, 0, 0, 0};
#if defined(__clang_major__) && (__clang_major__ >= 23)
      tdm8i gz8 = {0, 0, 0, 0, 0, 0, 0, 0};
      __builtin_amdgcn_tensor_load_to_lds(g0, g1, gz, gz, gz8, 0);
#else
      __builtin_amdgcn_tensor_load_to_lds(g0, g1, gz, gz, 0);
#endif
      __builtin_amdgcn_s_wait_tensorcnt(0);
    }
#else
    for (int tt = threadIdx.x; tt < 2048; tt += 256)
      zsh4[tt] = ((const uint4*)src)[tt];
#endif
    __syncthreads();
    for (int k0 = 0; k0 < 128; k0 += 32) {
      v16h a, bm;
      int hrow = lane & 15;
      if (hrow < NH) {
        const _Float16* ap = aatt + ((size_t)hrow*N_RES + i)*N_RES + c*128 + k0 + ((lane >> 4) << 3);
        __builtin_prefetch(ap + 128);
#pragma unroll
        for (int e = 0; e < 8; ++e) { a[e] = ap[e]; a[e+8] = ap[e+16]; }
      } else {
#pragma unroll
        for (int e = 0; e < 16; ++e) a[e] = (_Float16)0.f;
      }
      ld_k32_strided(zsh + (size_t)k0*CZ + n, CZ, lane, bm);
      acc = wmma32(a, bm, acc);
    }
  }
  int rb = (lane < 16) ? 0 : 8;
#pragma unroll
  for (int r = 0; r < 8; ++r) {
    int hh = rb + r;
    if (hh < NH) opair[((size_t)i*NH + hh)*CZ + n] = acc[r];
  }
}

// ---------------------------------------------------------------- concat assembly (o | fl_x | fl_y | fl_z | norm | o_pair)
__global__ void k_finalize(const float* __restrict__ obuf, const float* __restrict__ optbuf,
    const float* __restrict__ opair, const float* __restrict__ quats,
    const float* __restrict__ trans, _Float16* __restrict__ cat) {
  int idx = blockIdx.x * blockDim.x + threadIdx.x;
  if (idx >= N_RES * NH) return;
  int i = idx / NH, h = idx % NH;
  float q[4] = {quats[i*4], quats[i*4+1], quats[i*4+2], quats[i*4+3]};
  float R[3][3]; quat_to_rot(q, R);
  float t[3] = {trans[i*3], trans[i*3+1], trans[i*3+2]};
  _Float16* cr = cat + (size_t)i * CATD;
  for (int c = 0; c < CH; ++c) cr[h*CH + c] = (_Float16)obuf[(size_t)i*(NH*CH) + h*CH + c];
  for (int p = 0; p < NPV; ++p) {
    float gp[3], loc[3];
    for (int x = 0; x < 3; ++x) gp[x] = optbuf[(((size_t)i*NH + h)*NPV + p)*3 + x] - t[x];
    for (int x = 0; x < 3; ++x) loc[x] = R[0][x]*gp[0] + R[1][x]*gp[1] + R[2][x]*gp[2];
    cr[192 + h*NPV + p] = (_Float16)loc[0];
    cr[288 + h*NPV + p] = (_Float16)loc[1];
    cr[384 + h*NPV + p] = (_Float16)loc[2];
    cr[480 + h*NPV + p] = (_Float16)sqrtf(loc[0]*loc[0]+loc[1]*loc[1]+loc[2]*loc[2] + 1e-8f);
  }
  for (int c = 0; c < CZ; ++c) cr[576 + h*CZ + c] = (_Float16)opair[((size_t)i*NH + h)*CZ + c];
}

// ---------------------------------------------------------------- backbone update + frame emit
__global__ void k_bb(const float* __restrict__ s, const float* __restrict__ wbb,
    const float* __restrict__ bbb, float* __restrict__ quats, float* __restrict__ trans,
    float* __restrict__ frames) {
  int i = blockIdx.x * blockDim.x + threadIdx.x;
  if (i >= N_RES) return;
  const float* sr = s + (size_t)i * CS;
  float u[6];
  for (int o = 0; o < 6; ++o) {
    float acc = bbb[o];
    const float* w = wbb + o*CS;
    for (int k = 0; k < CS; ++k) acc += sr[k]*w[k];
    u[o] = acc;
  }
  float q[4] = {quats[i*4], quats[i*4+1], quats[i*4+2], quats[i*4+3]};
  float R[3][3]; quat_to_rot(q, R);            // rot from pre-update quats
  float rw = 1.f, rx = u[0], ry = u[1], rz = u[2];
  float nq[4];
  nq[0] = q[0]*rw - q[1]*rx - q[2]*ry - q[3]*rz;
  nq[1] = q[0]*rx + q[1]*rw + q[2]*rz - q[3]*ry;
  nq[2] = q[0]*ry - q[1]*rz + q[2]*rw + q[3]*rx;
  nq[3] = q[0]*rz + q[1]*ry - q[2]*rx + q[3]*rw;
  float inv = rsqrtf(nq[0]*nq[0]+nq[1]*nq[1]+nq[2]*nq[2]+nq[3]*nq[3] + 1e-30f);
  for (int x = 0; x < 4; ++x) nq[x] *= inv;
  float nt[3];
  for (int x = 0; x < 3; ++x)
    nt[x] = trans[i*3+x] + R[x][0]*u[3] + R[x][1]*u[4] + R[x][2]*u[5];
  for (int x = 0; x < 4; ++x) quats[i*4+x] = nq[x];
  for (int x = 0; x < 3; ++x) trans[i*3+x] = nt[x];
  float* fo = frames + (size_t)i * 7;
  fo[0] = nq[0]; fo[1] = nq[1]; fo[2] = nq[2]; fo[3] = nq[3];
  fo[4] = nt[0]*10.f; fo[5] = nt[1]*10.f; fo[6] = nt[2]*10.f;
}

// ================================================================ host launcher
extern "C" void kernel_launch(void* const* d_in, const int* in_sizes, int n_in,
                              void* d_out, int out_size, void* d_ws, size_t ws_size,
                              hipStream_t stream) {
  const float* s_in   = (const float*)d_in[0];
  const float* z_in   = (const float*)d_in[1];
  const float* rq     = (const float*)d_in[2];
  const float* rt     = (const float*)d_in[3];
  const float* mask   = (const float*)d_in[4];
  const float* ln_s_g = (const float*)d_in[5];
  const float* ln_s_b = (const float*)d_in[6];
  const float* ln_z_g = (const float*)d_in[7];
  const float* ln_z_b = (const float*)d_in[8];
  const float* w_in   = (const float*)d_in[9];
  const float* b_in   = (const float*)d_in[10];
  const float* w_q    = (const float*)d_in[11];
  const float* b_q    = (const float*)d_in[12];
  const float* w_kv   = (const float*)d_in[13];
  const float* b_kv   = (const float*)d_in[14];
  const float* w_qp   = (const float*)d_in[15];
  const float* b_qp   = (const float*)d_in[16];
  const float* w_kvp  = (const float*)d_in[17];
  const float* b_kvp  = (const float*)d_in[18];
  const float* w_b    = (const float*)d_in[19];
  const float* b_b    = (const float*)d_in[20];
  const float* head_w = (const float*)d_in[21];
  const float* w_o    = (const float*)d_in[22];
  const float* b_o    = (const float*)d_in[23];
  const float* ln_i_g = (const float*)d_in[24];
  const float* ln_i_b = (const float*)d_in[25];
  const float* ln_t_g = (const float*)d_in[26];
  const float* ln_t_b = (const float*)d_in[27];
  const float* w_t1   = (const float*)d_in[28];
  const float* b_t1   = (const float*)d_in[29];
  const float* w_t2   = (const float*)d_in[30];
  const float* b_t2   = (const float*)d_in[31];
  const float* w_t3   = (const float*)d_in[32];
  const float* b_t3   = (const float*)d_in[33];
  const float* w_bb   = (const float*)d_in[34];
  const float* b_bb   = (const float*)d_in[35];

  char* ws = (char*)d_ws;
  size_t off = 0;
  auto alloc = [&](size_t bytes) -> void* {
    void* p = ws + off;
    off = (off + bytes + 255) & ~(size_t)255;
    return p;
  };
  _Float16* Z16    = (_Float16*)alloc((size_t)N_RES*N_RES*CZ*2);
  float*    BPAIR  = (float*)   alloc((size_t)NH*N_RES*N_RES*4);
  _Float16* AATT   = (_Float16*)alloc((size_t)NH*N_RES*N_RES*2);
  float*    SCUR   = (float*)   alloc((size_t)N_RES*CS*4);
  _Float16* S16    = (_Float16*)alloc((size_t)N_RES*CS*2);
  _Float16* WIN16  = (_Float16*)alloc((size_t)CS*CS*2);
  _Float16* WPROJ  = (_Float16*)alloc((size_t)PROJD*CS*2);
  float*    BPROJ  = (float*)   alloc((size_t)PROJD*4);
  _Float16* WO16   = (_Float16*)alloc((size_t)CS*CATD*2);
  _Float16* WT1    = (_Float16*)alloc((size_t)CS*CS*2);
  _Float16* WT2    = (_Float16*)alloc((size_t)CS*CS*2);
  _Float16* WT3    = (_Float16*)alloc((size_t)CS*CS*2);
  float*    PROJ   = (float*)   alloc((size_t)N_RES*PROJD*4);
  _Float16* QHAT   = (_Float16*)alloc((size_t)NH*N_RES*32*2);
  _Float16* KHAT   = (_Float16*)alloc((size_t)NH*N_RES*32*2);
  float*    SSQ    = (float*)   alloc((size_t)NH*N_RES*4);
  float*    SSK    = (float*)   alloc((size_t)NH*N_RES*4);
  _Float16* VCAT   = (_Float16*)alloc((size_t)NH*N_RES*48*2);
  float*    OBUF   = (float*)   alloc((size_t)N_RES*NH*CH*4);
  float*    OPT    = (float*)   alloc((size_t)N_RES*NH*NPV*3*4);
  float*    OPAIR  = (float*)   alloc((size_t)N_RES*NH*CZ*4);
  _Float16* CAT16  = (_Float16*)alloc((size_t)N_RES*CATD*2);
  _Float16* T116   = (_Float16*)alloc((size_t)N_RES*CS*2);
  _Float16* T216   = (_Float16*)alloc((size_t)N_RES*CS*2);
  float*    QUATS  = (float*)   alloc((size_t)N_RES*4*4);
  float*    TRANS  = (float*)   alloc((size_t)N_RES*3*4);
  (void)ws_size; (void)n_in; (void)in_sizes; (void)out_size;

  auto cvt = [&](const float* src, _Float16* dst, int n) {
    k_f32_to_f16<<<(n + 255)/256, 256, 0, stream>>>(src, dst, n);
  };
  // weight conversion / packing (deterministic, cheap)
  cvt(w_in, WIN16, CS*CS);
  cvt(w_q,   WPROJ + 0,              192*CS);
  cvt(w_kv,  WPROJ + (size_t)192*CS, 384*CS);
  cvt(w_qp,  WPROJ + (size_t)576*CS, 144*CS);
  cvt(w_kvp, WPROJ + (size_t)720*CS, 432*CS);
  k_copy_f32<<<1, 256, 0, stream>>>(b_q,   BPROJ + 0,   192);
  k_copy_f32<<<2, 256, 0, stream>>>(b_kv,  BPROJ + 192, 384);
  k_copy_f32<<<1, 256, 0, stream>>>(b_qp,  BPROJ + 576, 144);
  k_copy_f32<<<2, 256, 0, stream>>>(b_kvp, BPROJ + 720, 432);
  cvt(w_o, WO16, CS*CATD);
  cvt(w_t1, WT1, CS*CS); cvt(w_t2, WT2, CS*CS); cvt(w_t3, WT3, CS*CS);
  k_init_rigid<<<2, 256, 0, stream>>>(rq, rt, QUATS, TRANS);

  // LN(z) + pair bias (block-invariant, once)
  k_lnz_bpair<<<N_RES*N_RES/8, 256, 0, stream>>>(z_in, ln_z_g, ln_z_b, w_b, b_b, Z16, BPAIR);
  // s = LN(s) @ w_in^T + b_in
  k_ln_rows<<<N_RES/8, 256, 0, stream>>>(s_in, ln_s_g, ln_s_b, nullptr, T116);
  k_gemm<<<dim3(3, 32), 256, 0, stream>>>(T116, CS, WIN16, CS, b_in, nullptr, SCUR, S16, CS,
                                          N_RES, CS, CS, 0);

  for (int blk = 0; blk < NBLK; ++blk) {
    // fused projections: [q | kv | q_pts | kv_pts]
    k_gemm<<<dim3(9, 32), 256, 0, stream>>>(S16, CS, WPROJ, CS, BPROJ, nullptr, PROJ, nullptr,
                                            PROJD, N_RES, PROJD, CS, 0);
    k_points<<<24, 256, 0, stream>>>(PROJ, QUATS, TRANS, head_w, QHAT, KHAT, VCAT, SSQ, SSK);
    k_attn<<<dim3(32, NH), 256, 0, stream>>>(QHAT, KHAT, SSQ, SSK, BPAIR, mask, VCAT,
                                             AATT, OBUF, OPT);
    k_opair<<<N_RES, 256, 0, stream>>>(AATT, Z16, OPAIR);
    k_finalize<<<24, 256, 0, stream>>>(OBUF, OPT, OPAIR, QUATS, TRANS, CAT16);
    // s = s + cat @ w_o^T + b_o ; s = LN_ipa(s)
    k_gemm<<<dim3(3, 32), 256, 0, stream>>>(CAT16, CATD, WO16, CATD, b_o, SCUR, SCUR, nullptr,
                                            CS, N_RES, CS, CATD, 0);
    k_ln_rows<<<N_RES/8, 256, 0, stream>>>(SCUR, ln_i_g, ln_i_b, SCUR, S16);
    // transition MLP
    k_gemm<<<dim3(3, 32), 256, 0, stream>>>(S16, CS, WT1, CS, b_t1, nullptr, nullptr, T116,
                                            CS, N_RES, CS, CS, 1);
    k_gemm<<<dim3(3, 32), 256, 0, stream>>>(T116, CS, WT2, CS, b_t2, nullptr, nullptr, T216,
                                            CS, N_RES, CS, CS, 1);
    k_gemm<<<dim3(3, 32), 256, 0, stream>>>(T216, CS, WT3, CS, b_t3, SCUR, SCUR, nullptr,
                                            CS, N_RES, CS, CS, 0);
    k_ln_rows<<<N_RES/8, 256, 0, stream>>>(SCUR, ln_t_g, ln_t_b, SCUR, S16);
    // backbone frame update
    float* frame_out = (float*)d_out + (size_t)blk * N_RES * 7;
    k_bb<<<2, 256, 0, stream>>>(SCUR, w_bb, b_bb, QUATS, TRANS, frame_out);
  }
  // final s output after frames
  k_copy_f32<<<(N_RES*CS + 255)/256, 256, 0, stream>>>(
      SCUR, (float*)d_out + (size_t)NBLK*N_RES*7, N_RES*CS);
}
